// Graph2Edits_84447646974091
// MI455X (gfx1250) — compile-verified, matrix operand order checked by reading, not run on hardware
//
#include <hip/hip_runtime.h>
#include <stdint.h>

// ---------------------------------------------------------------------------
// Problem dimensions (match reference)
// ---------------------------------------------------------------------------
constexpr int kAtoms   = 65536;
constexpr int kBonds   = 131072;
constexpr int kUBonds  = 65536;
constexpr int kMaxNb   = 6;
constexpr int kAtomF   = 98;
constexpr int kBondF   = 104;
constexpr int kH       = 256;
constexpr int kMLP     = 512;
constexpr int kAtomOut = 33;
constexpr int kBondOut = 24;
constexpr int kMols    = 2048;
constexpr int kDepth   = 6;

typedef _Float16 h16;
typedef __attribute__((ext_vector_type(16))) _Float16 v16h;
typedef __attribute__((ext_vector_type(8)))  _Float16 v8h;
typedef __attribute__((ext_vector_type(4)))  _Float16 v4h;
typedef __attribute__((ext_vector_type(8)))  float    v8f;
typedef int v4i_gcc __attribute__((vector_size(16)));   // builtin's b128 unit

// CDNA5 async global->LDS path (ASYNCcnt-tracked DMA, no VGPR round trip).
#if __has_builtin(__builtin_amdgcn_global_load_async_to_lds_b128) && \
    __has_builtin(__builtin_amdgcn_s_wait_asynccnt)
#define ASYNC_LDS 1
#else
#define ASYNC_LDS 0
#endif

// ---------------------------------------------------------------------------
// Weight transpose + f32 -> f16 convert:  W[K][N] -> Wt[N][Kpad], zero pad
// ---------------------------------------------------------------------------
__global__ void wt_convert(const float* __restrict__ W, h16* __restrict__ Wt,
                           int K, int N, int Kpad) {
  int idx = blockIdx.x * blockDim.x + threadIdx.x;
  if (idx >= N * Kpad) return;
  int n = idx / Kpad, k = idx % Kpad;
  Wt[(size_t)n * Kpad + k] = (k < K) ? (h16)W[(size_t)k * N + n] : (h16)0.0f;
}

// Row-major f32 [M][K] -> f16 [M][Kpad], zero pad
__global__ void pad_convert(const float* __restrict__ X, h16* __restrict__ Y,
                            int M, int K, int Kpad) {
  int idx = blockIdx.x * blockDim.x + threadIdx.x;
  if (idx >= M * Kpad) return;
  int m = idx / Kpad, k = idx % Kpad;
  Y[(size_t)m * Kpad + k] = (k < K) ? (h16)X[(size_t)m * K + k] : (h16)0.0f;
}

// ---------------------------------------------------------------------------
// Tiled WMMA GEMM:  out[M][N] = act( A[M][K] * Bt[N][K]^T + bias + addC )
//   A, Bt f16, K multiple of 64; f32 accum via v_wmma_f32_16x16x32_f16.
//   Block tile 128x128, 256 threads = 8 waves, wave tile 32x64 (2x4 frags).
//   Double-buffered LDS staging (BK=64 -> 2 k-steps / 16 WMMAs per stage).
// ---------------------------------------------------------------------------
#define LDSW 72   // f16 row stride for 64-wide tile (+8 pad; 144B, 16B aligned)

__global__ __launch_bounds__(256)
void gemm_wmma(const h16* __restrict__ A, const h16* __restrict__ Bt,
               const float* __restrict__ bias, const float* __restrict__ addC,
               float* __restrict__ out, h16* __restrict__ out_h,
               float* __restrict__ out_raw,
               int M, int N, int K, int do_relu) {
  __shared__ h16 sA[2][128 * LDSW];
  __shared__ h16 sB[2][128 * LDSW];

  const int tid   = threadIdx.x;
  const int lane  = tid & 31;
  const int wave  = tid >> 5;
  const int wm    = wave & 3;   // 0..3 -> 32-row strip
  const int wn    = wave >> 2;  // 0..1 -> 64-col strip
  const int lhalf = lane >> 4;  // K-half selector (ISA 16-bit A/B layout)
  const int lrow  = lane & 15;

  const int m0 = blockIdx.x * 128;
  const int n0 = blockIdx.y * 128;

  // Per-thread staging slots: 4 x 16B chunks per matrix per 128x64 stage.
  int srow[4], scol[4];
  const h16* gA[4];
  const h16* gB[4];
#pragma unroll
  for (int i = 0; i < 4; ++i) {
    int id  = tid + (i << 8);       // 0..1023
    srow[i] = id >> 3;              // 0..127
    scol[i] = (id & 7) << 3;        // 0,8,...,56 (f16)
    gA[i]   = A  + (size_t)(m0 + srow[i]) * K + scol[i];
    gB[i]   = Bt + (size_t)(n0 + srow[i]) * K + scol[i];
  }

  // Issue stage k0 -> buffer buf. Async path DMAs straight into LDS; the
  // fallback batches all 8 global loads into distinct regs (one loadcnt wait).
  auto issue_stage = [&](int k0, int buf, v8h* ta, v8h* tb) {
#if ASYNC_LDS
    (void)ta; (void)tb;
#pragma unroll
    for (int i = 0; i < 4; ++i) {
      __builtin_amdgcn_global_load_async_to_lds_b128(
          (v4i_gcc*)(uintptr_t)(gA[i] + k0),
          (__attribute__((address_space(3))) v4i_gcc*)
              (__attribute__((address_space(3))) void*)&sA[buf][srow[i] * LDSW + scol[i]],
          0, 0);
      __builtin_amdgcn_global_load_async_to_lds_b128(
          (v4i_gcc*)(uintptr_t)(gB[i] + k0),
          (__attribute__((address_space(3))) v4i_gcc*)
              (__attribute__((address_space(3))) void*)&sB[buf][srow[i] * LDSW + scol[i]],
          0, 0);
    }
#else
    (void)buf;
#pragma unroll
    for (int i = 0; i < 4; ++i) {
      ta[i] = *(const v8h*)(gA[i] + k0);
      tb[i] = *(const v8h*)(gB[i] + k0);
    }
#endif
  };
  auto commit_stage = [&](int buf, const v8h* ta, const v8h* tb) {
#if ASYNC_LDS
    (void)buf; (void)ta; (void)tb;
    __builtin_amdgcn_s_wait_asynccnt(0);
#else
#pragma unroll
    for (int i = 0; i < 4; ++i) {
      *(v8h*)&sA[buf][srow[i] * LDSW + scol[i]] = ta[i];
      *(v8h*)&sB[buf][srow[i] * LDSW + scol[i]] = tb[i];
    }
#endif
  };

  v8f acc[2][4];
  const v8f zero = {0.f, 0.f, 0.f, 0.f, 0.f, 0.f, 0.f, 0.f};
#pragma unroll
  for (int i = 0; i < 2; ++i)
#pragma unroll
    for (int j = 0; j < 4; ++j) acc[i][j] = zero;

  v8h ta[4], tb[4];
  issue_stage(0, 0, ta, tb);
  commit_stage(0, ta, tb);
  __syncthreads();

  const int nk = K >> 6;  // 128x64 stages
  for (int kc = 0; kc < nk; ++kc) {
    const int buf = kc & 1;
    v8h na[4], nb[4];
    if (kc + 1 < nk) issue_stage((kc + 1) << 6, buf ^ 1, na, nb);

    // Two 16x16x32 k-steps from the staged 64-wide tile.
#pragma unroll
    for (int s = 0; s < 2; ++s) {
      // ISA 16-bit fragment layout: lane = row (M for A, N for Bt);
      // elems 0..7 = K[lhalf*8..+7], elems 8..15 = K[16+lhalf*8..+7].
      v16h af[2], bf[4];
#pragma unroll
      for (int fm = 0; fm < 2; ++fm) {
        const h16* p = &sA[buf][(wm * 32 + fm * 16 + lrow) * LDSW + s * 32 + lhalf * 8];
        v8h lo = *(const v8h*)p;
        v8h hi = *(const v8h*)(p + 16);
        af[fm] = __builtin_shufflevector(lo, hi, 0,1,2,3,4,5,6,7,8,9,10,11,12,13,14,15);
      }
#pragma unroll
      for (int fn = 0; fn < 4; ++fn) {
        const h16* p = &sB[buf][(wn * 64 + fn * 16 + lrow) * LDSW + s * 32 + lhalf * 8];
        v8h lo = *(const v8h*)p;
        v8h hi = *(const v8h*)(p + 16);
        bf[fn] = __builtin_shufflevector(lo, hi, 0,1,2,3,4,5,6,7,8,9,10,11,12,13,14,15);
      }
#pragma unroll
      for (int fm = 0; fm < 2; ++fm)
#pragma unroll
        for (int fn = 0; fn < 4; ++fn)
          acc[fm][fn] = __builtin_amdgcn_wmma_f32_16x16x32_f16(
              false, af[fm], false, bf[fn], (short)0, acc[fm][fn], false, false);
    }

    if (kc + 1 < nk) commit_stage(buf ^ 1, na, nb);
    __syncthreads();
  }

  // Epilogue: C/D layout — VGPR r, lane L: row = r + 8*(L/16), col = L%16.
#pragma unroll
  for (int fm = 0; fm < 2; ++fm)
#pragma unroll
    for (int fn = 0; fn < 4; ++fn) {
      const int col = n0 + wn * 64 + fn * 16 + lrow;
      const int rb  = m0 + wm * 32 + fm * 16 + lhalf * 8;
      const float bc = bias ? bias[col] : 0.0f;
#pragma unroll
      for (int r = 0; r < 8; ++r) {
        const size_t o = (size_t)(rb + r) * N + col;
        float v = acc[fm][fn][r] + bc;
        if (addC)    v += addC[o];
        if (out_raw) out_raw[o] = v;
        if (do_relu) v = fmaxf(v, 0.0f);
        if (out)     out[o]   = v;
        if (out_h)   out_h[o] = (h16)v;
      }
    }
}

// ---------------------------------------------------------------------------
// a_msg[a] = sum_j msg[a2b[a][j]]            (f32, float4 chunks)
// ---------------------------------------------------------------------------
__global__ void gather_sum(const float* __restrict__ msg,
                           const int* __restrict__ a2b,
                           float* __restrict__ a_msg) {
  int idx = blockIdx.x * blockDim.x + threadIdx.x;   // kAtoms * 64
  if (idx >= kAtoms * (kH / 4)) return;
  int a = idx >> 6, c = (idx & 63) << 2;
  float s0 = 0.f, s1 = 0.f, s2 = 0.f, s3 = 0.f;
#pragma unroll
  for (int j = 0; j < kMaxNb; ++j) {
    const float* p = msg + (size_t)a2b[a * kMaxNb + j] * kH + c;
    s0 += p[0]; s1 += p[1]; s2 += p[2]; s3 += p[3];
  }
  float* q = a_msg + (size_t)a * kH + c;
  q[0] = s0; q[1] = s1; q[2] = s2; q[3] = s3;
}

// mdiff_f16[b] = (f16)(a_msg[b2a[b]] - msg[b2revb[b]])
__global__ void mdiff_f16(const float* __restrict__ a_msg,
                          const float* __restrict__ msg,
                          const int* __restrict__ b2a,
                          const int* __restrict__ b2revb,
                          h16* __restrict__ outh) {
  int idx = blockIdx.x * blockDim.x + threadIdx.x;   // kBonds * 64
  if (idx >= kBonds * (kH / 4)) return;
  int b = idx >> 6, c = (idx & 63) << 2;
  const float* pa = a_msg + (size_t)b2a[b]    * kH + c;
  const float* pm = msg   + (size_t)b2revb[b] * kH + c;
  v4h o;
  o[0] = (h16)(pa[0] - pm[0]); o[1] = (h16)(pa[1] - pm[1]);
  o[2] = (h16)(pa[2] - pm[2]); o[3] = (h16)(pa[3] - pm[3]);
  *(v4h*)(outh + (size_t)b * kH + c) = o;
}

// a_cat[a] = [ f_atoms[a] (98) | a_msg[a] (256) | zeros (30) ]  -> f16 [.][384]
__global__ void build_acat(const float* __restrict__ f_atoms,
                           const float* __restrict__ a_msg,
                           h16* __restrict__ outh) {
  int idx = blockIdx.x * blockDim.x + threadIdx.x;
  if (idx >= kAtoms * 384) return;
  int a = idx / 384, c = idx % 384;
  float v = 0.0f;
  if (c < kAtomF)            v = f_atoms[(size_t)a * kAtomF + c];
  else if (c < kAtomF + kH)  v = a_msg[(size_t)a * kH + (c - kAtomF)];
  outh[(size_t)a * 384 + c] = (h16)v;
}

// bond_feats_f16[e] = [ atom_f16[i0] | atom_f16[i1] ]  (16B chunks)
__global__ void bond_gather(const h16* __restrict__ atom_h,
                            const int* __restrict__ bond_idx,
                            h16* __restrict__ outh) {
  int idx = blockIdx.x * blockDim.x + threadIdx.x;   // kUBonds * 64 chunks
  if (idx >= kUBonds * 64) return;
  int e = idx >> 6, q = idx & 63;
  int which = q >> 5;                 // 0: endpoint 0, 1: endpoint 1
  int off   = (q & 31) << 3;          // 8-half offset within 256
  int row   = bond_idx[e * 2 + which];
  *(v8h*)(outh + (size_t)e * (2 * kH) + which * kH + off) =
      *(const v8h*)(atom_h + (size_t)row * kH + off);
}

__global__ void zero_f32(float* __restrict__ p, int n) {
  int i = blockIdx.x * blockDim.x + threadIdx.x;
  if (i < n) p[i] = 0.0f;
}

// graph_vecs[mol_ids[a]] += atom_feats[a]
__global__ void seg_sum(const float* __restrict__ atom_f,
                        const int* __restrict__ mol_ids,
                        float* __restrict__ gv) {
  int idx = blockIdx.x * blockDim.x + threadIdx.x;
  if (idx >= kAtoms * kH) return;
  int a = idx >> 8, c = idx & 255;
  atomicAdd(&gv[(size_t)mol_ids[a] * kH + c], atom_f[idx]);
}

// Small-N head: out[m][n] = sum_k Hrow[m][k]*W[k][n] + b[n]   (one block per row)
__global__ __launch_bounds__(64)
void gemm_smalln(const float* __restrict__ Hm, const float* __restrict__ W,
                 const float* __restrict__ b, float* __restrict__ out,
                 int K, int N) {
  __shared__ float row[512];
  const int m = blockIdx.x;
  for (int i = threadIdx.x; i < K; i += 64) row[i] = Hm[(size_t)m * K + i];
  __syncthreads();
  const int n = threadIdx.x;
  if (n >= N) return;
  float acc = b[n];
  for (int k = 0; k < K; ++k) acc += row[k] * W[(size_t)k * N + n];
  out[(size_t)m * N + n] = acc;
}

// ---------------------------------------------------------------------------
// Host orchestration
// ---------------------------------------------------------------------------
static inline int cdiv(long long a, int b) { return (int)((a + b - 1) / b); }

extern "C" void kernel_launch(void* const* d_in, const int* in_sizes, int n_in,
                              void* d_out, int out_size, void* d_ws, size_t ws_size,
                              hipStream_t stream) {
  (void)in_sizes; (void)n_in; (void)out_size; (void)ws_size;
  const float* f_atoms = (const float*)d_in[0];
  const float* f_bonds = (const float*)d_in[1];
  const int*   a2b     = (const int*)d_in[2];
  const int*   b2a     = (const int*)d_in[3];
  const int*   b2revb  = (const int*)d_in[4];
  const int*   bondIdx = (const int*)d_in[5];
  const int*   mol_ids = (const int*)d_in[6];
  const float* prev_h  = (const float*)d_in[7];
  const float* Wi  = (const float*)d_in[8];
  const float* Wh  = (const float*)d_in[9];
  const float* Wo  = (const float*)d_in[10];
  const float* bo  = (const float*)d_in[11];
  const float* Wvv = (const float*)d_in[12];
  const float* Wvc = (const float*)d_in[13];
  const float* Wa1 = (const float*)d_in[14];
  const float* ba1 = (const float*)d_in[15];
  const float* Wa2 = (const float*)d_in[16];
  const float* ba2 = (const float*)d_in[17];
  const float* Wb1 = (const float*)d_in[18];
  const float* bb1 = (const float*)d_in[19];
  const float* Wb2 = (const float*)d_in[20];
  const float* bb2 = (const float*)d_in[21];
  const float* Wg1 = (const float*)d_in[22];
  const float* bg1 = (const float*)d_in[23];
  const float* Wg2 = (const float*)d_in[24];
  const float* bg2 = (const float*)d_in[25];
  float* outp = (float*)d_out;

  // ---- workspace carve-up (256B aligned) ----
  char* ws = (char*)d_ws;
  size_t off = 0;
  auto take = [&](size_t bytes) -> void* {
    void* p = ws + off;
    off = (off + bytes + 255) & ~(size_t)255;
    return p;
  };
  h16* Wi_t  = (h16*)take((size_t)256 * 128 * 2);
  h16* Wh_t  = (h16*)take((size_t)256 * 256 * 2);
  h16* Wo_t  = (h16*)take((size_t)256 * 384 * 2);
  h16* Wvv_t = (h16*)take((size_t)256 * 256 * 2);
  h16* Wvc_t = (h16*)take((size_t)256 * 256 * 2);
  h16* Wa1_t = (h16*)take((size_t)512 * 256 * 2);
  h16* Wb1_t = (h16*)take((size_t)512 * 512 * 2);
  h16* Wg1_t = (h16*)take((size_t)512 * 256 * 2);
  float* inp   = (float*)take((size_t)kBonds * kH * 4);   // reused: t1, MLP hidden
  float* msg   = (float*)take((size_t)kBonds * kH * 4);
  float* a_msg = (float*)take((size_t)kAtoms * kH * 4);
  h16*   big16 = (h16*)take((size_t)kBonds * kH * 2);     // A0 / mdiff / a_cat / bond_feats
  h16*   h16a  = (h16*)take((size_t)kAtoms * kH * 2);     // a_feats f16
  h16*   h16b  = (h16*)take((size_t)kAtoms * kH * 2);     // prev_h f16 -> atom_feats f16
  float* atomf = (float*)take((size_t)kAtoms * kH * 4);   // atom_feats f32
  float* gv    = (float*)take((size_t)kMols * kH * 4);
  h16*   gv16  = (h16*)take((size_t)kMols * kH * 2);
  float* hg    = (float*)take((size_t)kMols * kMLP * 4);

  float* t1     = inp;  // prev_h @ Wvv (after MPN loop, inp region is free)
  float* hidden = inp;  // MLP hidden activations

  const int T = 256;
  auto g1 = [&](long long n) { return dim3(cdiv(n, T)); };

  // ---- weight prep (transpose + f16, K padded to mult of 64) ----
  wt_convert<<<g1(256 * 128), T, 0, stream>>>(Wi,  Wi_t,  104, 256, 128);
  wt_convert<<<g1(256 * 256), T, 0, stream>>>(Wh,  Wh_t,  256, 256, 256);
  wt_convert<<<g1(256 * 384), T, 0, stream>>>(Wo,  Wo_t,  354, 256, 384);
  wt_convert<<<g1(256 * 256), T, 0, stream>>>(Wvv, Wvv_t, 256, 256, 256);
  wt_convert<<<g1(256 * 256), T, 0, stream>>>(Wvc, Wvc_t, 256, 256, 256);
  wt_convert<<<g1(512 * 256), T, 0, stream>>>(Wa1, Wa1_t, 256, 512, 256);
  wt_convert<<<g1(512 * 512), T, 0, stream>>>(Wb1, Wb1_t, 512, 512, 512);
  wt_convert<<<g1(512 * 256), T, 0, stream>>>(Wg1, Wg1_t, 256, 512, 256);

  // ---- MPN: inp = f_bonds @ Wi ; msg = relu(inp) ----
  pad_convert<<<g1((long long)kBonds * 128), T, 0, stream>>>(f_bonds, big16, kBonds, kBondF, 128);
  gemm_wmma<<<dim3(kBonds / 128, 256 / 128), T, 0, stream>>>(
      big16, Wi_t, nullptr, nullptr, msg, nullptr, inp, kBonds, 256, 128, 1);

  // ---- depth-1 message passing steps ----
  for (int d = 0; d < kDepth - 1; ++d) {
    gather_sum<<<g1((long long)kAtoms * 64), T, 0, stream>>>(msg, a2b, a_msg);
    mdiff_f16<<<g1((long long)kBonds * 64), T, 0, stream>>>(a_msg, msg, b2a, b2revb, big16);
    gemm_wmma<<<dim3(kBonds / 128, 256 / 128), T, 0, stream>>>(
        big16, Wh_t, nullptr, inp, msg, nullptr, nullptr, kBonds, 256, 256, 1);
  }

  // ---- readout: a_feats = relu([f_atoms | a_msg] @ Wo + bo)  (f16 out) ----
  gather_sum<<<g1((long long)kAtoms * 64), T, 0, stream>>>(msg, a2b, a_msg);
  build_acat<<<g1((long long)kAtoms * 384), T, 0, stream>>>(f_atoms, a_msg, big16);
  gemm_wmma<<<dim3(kAtoms / 128, 256 / 128), T, 0, stream>>>(
      big16, Wo_t, bo, nullptr, nullptr, h16a, nullptr, kAtoms, 256, 384, 1);

  // ---- state update: atom_feats = relu(prev_h@Wvv + a_feats@Wvc) ----
  pad_convert<<<g1((long long)kAtoms * 256), T, 0, stream>>>(prev_h, h16b, kAtoms, 256, 256);
  gemm_wmma<<<dim3(kAtoms / 128, 256 / 128), T, 0, stream>>>(
      h16b, Wvv_t, nullptr, nullptr, t1, nullptr, nullptr, kAtoms, 256, 256, 0);
  gemm_wmma<<<dim3(kAtoms / 128, 256 / 128), T, 0, stream>>>(
      h16a, Wvc_t, nullptr, t1, atomf, h16b, nullptr, kAtoms, 256, 256, 1);

  // ---- bond features + per-molecule readout ----
  bond_gather<<<g1((long long)kUBonds * 64), T, 0, stream>>>(h16b, bondIdx, big16);
  zero_f32<<<g1((long long)kMols * kH), T, 0, stream>>>(gv, kMols * kH);
  seg_sum<<<g1((long long)kAtoms * kH), T, 0, stream>>>(atomf, mol_ids, gv);

  // ---- heads (output concat: bond | atom | graph) ----
  float* out_bond  = outp;
  float* out_atom  = outp + (size_t)kUBonds * kBondOut;       // 1572864
  float* out_graph = out_atom + (size_t)kAtoms * kAtomOut;    // 3735552

  // atom head
  gemm_wmma<<<dim3(kAtoms / 128, kMLP / 128), T, 0, stream>>>(
      h16b, Wa1_t, ba1, nullptr, hidden, nullptr, nullptr, kAtoms, kMLP, 256, 1);
  gemm_smalln<<<dim3(kAtoms), 64, 0, stream>>>(hidden, Wa2, ba2, out_atom, kMLP, kAtomOut);

  // bond head
  gemm_wmma<<<dim3(kUBonds / 128, kMLP / 128), T, 0, stream>>>(
      big16, Wb1_t, bb1, nullptr, hidden, nullptr, nullptr, kUBonds, kMLP, 512, 1);
  gemm_smalln<<<dim3(kUBonds), 64, 0, stream>>>(hidden, Wb2, bb2, out_bond, kMLP, kBondOut);

  // graph head
  pad_convert<<<g1((long long)kMols * 256), T, 0, stream>>>(gv, gv16, kMols, 256, 256);
  gemm_wmma<<<dim3(kMols / 128, kMLP / 128), T, 0, stream>>>(
      gv16, Wg1_t, bg1, nullptr, hg, nullptr, nullptr, kMols, kMLP, 256, 1);
  gemm_smalln<<<dim3(kMols), 64, 0, stream>>>(hg, Wg2, bg2, out_graph, kMLP, 1);
}
